// GroupedQueryAttention_90555090469434
// MI455X (gfx1250) — matrix operand order repaired and checked by reading, not compiled, not run on hardware
//
#include <hip/hip_runtime.h>
#include <hip/hip_bf16.h>

// ---------------------------------------------------------------------------
// GQA layer for MI455X (gfx1250): bf16 WMMA GEMMs + flash attention,
// with CDNA5 async global->LDS staging (ASYNCcnt) and double-buffered tiles.
// Dims fixed by the reference: B=1, S=2048, D=2048, H=32, G=8, HD=64.
// ---------------------------------------------------------------------------

#define S_DIM 2048
#define D_DIM 2048
#define H_NUM 32
#define G_NUM 8
#define HD_DIM 64
#define HQ_DIM (H_NUM * HD_DIM)  // 2048
#define HK_DIM (G_NUM * HD_DIM)  // 512

typedef __attribute__((ext_vector_type(16))) __bf16 bf16x16;
typedef __attribute__((ext_vector_type(8)))  float  f32x8;
typedef __attribute__((ext_vector_type(4)))  int    v4i;

union FragU {
  bf16x16 v;
  uint4   u[2];
};

// ---- CDNA5 async global->LDS copy (ASYNCcnt-tracked) ----------------------
#if defined(__AMDGCN__)
#if __has_builtin(__builtin_amdgcn_global_load_async_to_lds_b128)
typedef __attribute__((address_space(1))) v4i* async_gp_t;  // global int4*
typedef __attribute__((address_space(3))) v4i* async_lp_t;  // LDS int4*
__device__ __forceinline__ void async_g2l_b128(const unsigned short* g, unsigned short* l) {
  __builtin_amdgcn_global_load_async_to_lds_b128(
      (async_gp_t)g, (async_lp_t)l, 0, 0);
}
#else
__device__ __forceinline__ void async_g2l_b128(const unsigned short* g, unsigned short* l) {
  unsigned loff = (unsigned)(unsigned long long)l;  // LDS aperture: low 32 bits = LDS offset
  asm volatile("global_load_async_to_lds_b128 %0, %1, off"
               :: "v"(loff), "v"(g) : "memory");
}
#endif
#if __has_builtin(__builtin_amdgcn_s_wait_asynccnt)
#define WAIT_ASYNCCNT(N) __builtin_amdgcn_s_wait_asynccnt(N)
#else
#define WAIT_ASYNCCNT(N) asm volatile("s_wait_asynccnt %0" :: "i"(N) : "memory")
#endif
#else
// Host pass: never executed; keep it compilable.
__device__ __forceinline__ void async_g2l_b128(const unsigned short* g, unsigned short* l) {
  *(uint4*)l = *(const uint4*)g;
}
#define WAIT_ASYNCCNT(N) ((void)0)
#endif

__device__ __forceinline__ unsigned short f2bf(float f) {
  unsigned int u = __float_as_uint(f);
  unsigned int r = u + 0x7FFFu + ((u >> 16) & 1u);  // round-to-nearest-even
  return (unsigned short)(r >> 16);
}

__device__ __forceinline__ f32x8 wmma_bf16(const FragU& a, const FragU& b, f32x8 c) {
  return __builtin_amdgcn_wmma_f32_16x16x32_bf16(
      /*neg_a=*/false, a.v, /*neg_b=*/false, b.v,
      /*c_mod=*/(short)0, c, /*reuse_a=*/false, /*reuse_b=*/false);
}

// A-matrix fragment (16x32 bf16): lane holds row = lane%16;
// per-lane K runs: {kb..kb+7, 16+kb..16+kb+7}, kb = 8*(lane/16).
__device__ __forceinline__ void load_fragA(FragU& f, const unsigned short* rowp, int lhalf) {
  const unsigned short* p = rowp + lhalf * 8;
  f.u[0] = *(const uint4*)(p);
  f.u[1] = *(const uint4*)(p + 16);
}

// B-matrix fragment (32x16 bf16) from row-major N x K storage (B[k][n] = W[n][k]):
// lane holds column n = lane%16; per-lane K runs: {kb..kb+7, 16+kb..16+kb+7}, kb = 8*(lane/16).
__device__ __forceinline__ void load_fragB(FragU& f, const unsigned short* rowp, int lhalf) {
  const unsigned short* p = rowp + lhalf * 8;
  f.u[0] = *(const uint4*)(p);
  f.u[1] = *(const uint4*)(p + 16);
}

// ---------------------------------------------------------------------------
// fp32 -> bf16 conversion (grid-stride)
// ---------------------------------------------------------------------------
__global__ __launch_bounds__(256) void k_f32_to_bf16(const float* __restrict__ in,
                                                     unsigned short* __restrict__ out,
                                                     int n) {
  for (int i = blockIdx.x * blockDim.x + threadIdx.x; i < n; i += gridDim.x * blockDim.x)
    out[i] = f2bf(in[i]);
}

// ---------------------------------------------------------------------------
// bf16 GEMM: C[M,N] (fp32) = A[M,K] (bf16) @ W[N,K]^T (bf16)
// Block tile 128x128, BK=32; 256 threads = 8 waves (2x4), each wave computes
// a 4x2 grid of 16x16 WMMA tiles (8 WMMA / K-step).  LDS tiles are double-
// buffered and filled with async global->LDS DMA; s_wait_asynccnt<=4 keeps
// exactly the next tile's 4 per-thread transfers in flight during compute.
// ---------------------------------------------------------------------------
__global__ __launch_bounds__(256) void k_gemm_bf16_nt(const unsigned short* __restrict__ A,
                                                      const unsigned short* __restrict__ W,
                                                      float* __restrict__ C,
                                                      int M, int N, int K) {
  __shared__ __align__(16) unsigned short As[2][128 * 32];
  __shared__ __align__(16) unsigned short Bs[2][128 * 32];

  const int tid   = threadIdx.x;
  const int lane  = tid & 31;
  const int wave  = tid >> 5;
  const int wr    = wave >> 2;       // 0..1 (M, 64 rows each)
  const int wc    = wave & 3;        // 0..3 (N, 32 cols each)
  const int lrow  = lane & 15;
  const int lhalf = lane >> 4;
  const int m0 = blockIdx.y * 128;
  const int n0 = blockIdx.x * 128;

  f32x8 acc[4][2] = {};

  // 128x32 tile = 512 uint4; 256 threads -> 2 chunks each per matrix.
  auto issue_tile = [&](int k0, int buf) {
#pragma unroll
    for (int i = 0; i < 2; ++i) {
      int idx = tid + i * 256;         // 0..511
      int r = idx >> 2, c = idx & 3;   // row 0..127, 8-elem group 0..3
      async_g2l_b128(&A[(size_t)(m0 + r) * K + k0 + c * 8], &As[buf][r * 32 + c * 8]);
      async_g2l_b128(&W[(size_t)(n0 + r) * K + k0 + c * 8], &Bs[buf][r * 32 + c * 8]);
    }
  };

  issue_tile(0, 0);

  int it = 0;
  for (int k0 = 0; k0 < K; k0 += 32, ++it) {
    const int buf = it & 1;
    if (k0 + 32 < K) {
      issue_tile(k0 + 32, buf ^ 1);   // prefetch next tile into other buffer
      WAIT_ASYNCCNT(4);               // current tile (4 per-thread ops) landed
    } else {
      WAIT_ASYNCCNT(0);
    }
    __syncthreads();

    FragU a[4], b[2];
#pragma unroll
    for (int mi = 0; mi < 4; ++mi)
      load_fragA(a[mi], &As[buf][(wr * 64 + mi * 16 + lrow) * 32], lhalf);
#pragma unroll
    for (int ni = 0; ni < 2; ++ni)
      load_fragB(b[ni], &Bs[buf][(wc * 32 + ni * 16 + lrow) * 32], lhalf);

#pragma unroll
    for (int mi = 0; mi < 4; ++mi)
#pragma unroll
      for (int ni = 0; ni < 2; ++ni)
        acc[mi][ni] = wmma_bf16(a[mi], b[ni], acc[mi][ni]);
    __syncthreads();  // protect buf from next iteration's DMA writes
  }

  // Epilogue: C layout -> row = vgpr + 8*lhalf, col = lrow.
#pragma unroll
  for (int mi = 0; mi < 4; ++mi)
#pragma unroll
    for (int ni = 0; ni < 2; ++ni)
#pragma unroll
      for (int r = 0; r < 8; ++r) {
        int row = m0 + wr * 64 + mi * 16 + r + 8 * lhalf;
        int col = n0 + wc * 32 + ni * 16 + lrow;
        C[(size_t)row * N + col] = acc[mi][ni][r];
      }
}

// ---------------------------------------------------------------------------
// RMSNorm + RoPE + transpose (s,h,hd) -> (h,s,hd), emit bf16.
// One block = one (s, h) row of 64; 64 threads (2 waves).
// doNR=0 -> plain convert+transpose (V path).
// ---------------------------------------------------------------------------
__global__ __launch_bounds__(64) void k_norm_rope_tr(const float* __restrict__ in,     // (S, Hn, 64)
                                                     unsigned short* __restrict__ out, // (Hn, S, 64)
                                                     const float* __restrict__ gamma,  // (64) or unused
                                                     const float* __restrict__ cosT,   // (S, 64)
                                                     const float* __restrict__ sinT,   // (S, 64)
                                                     int Hn, int S, int doNR) {
  __shared__ float wsum[2];
  __shared__ float yrow[64];

  const int s = blockIdx.x;
  const int h = blockIdx.y;
  const int d = threadIdx.x;

  float v = in[((size_t)s * Hn + h) * 64 + d];
  float outv;
  if (doNR) {
    float ss = v * v;
#pragma unroll
    for (int off = 16; off >= 1; off >>= 1) ss += __shfl_xor(ss, off, 32);
    if ((d & 31) == 0) wsum[d >> 5] = ss;
    __syncthreads();
    float tot = wsum[0] + wsum[1];
    float y = v * rsqrtf(tot * (1.0f / 64.0f) + 1e-6f) * gamma[d];
    yrow[d] = y;
    __syncthreads();
    float c = cosT[(size_t)s * 64 + d];
    float sn = sinT[(size_t)s * 64 + d];
    float partner = (d < 32) ? -yrow[d + 32] : yrow[d - 32];
    outv = y * c + partner * sn;
  } else {
    outv = v;
  }
  out[((size_t)h * S + s) * 64 + d] = f2bf(outv);
}

// ---------------------------------------------------------------------------
// Flash attention (causal, GQA): grid (S/64, H), 128 threads = 4 waves,
// each wave owns one 16-row q tile. Online softmax in fp32, WMMA bf16 for
// both q@k^T and p@v.  K tile staged with async DMA; V staged transposed.
// Output written bf16 to ctx (s, h*64+hd).
// ---------------------------------------------------------------------------
__global__ __launch_bounds__(128) void k_flash_attn(const unsigned short* __restrict__ Q,  // (H,S,64)
                                                    const unsigned short* __restrict__ Kk, // (G,S,64)
                                                    const unsigned short* __restrict__ V,  // (G,S,64)
                                                    unsigned short* __restrict__ O,        // (S, H*64)
                                                    int S, int ldo) {
  __shared__ __align__(16) unsigned short Ks[64 * 64];
  __shared__ __align__(16) unsigned short Vt[64 * 64];      // transposed: [hd][key]
  __shared__ __align__(16) unsigned short Ps[4][16 * 64];   // per-wave P tile

  const int h  = blockIdx.y;
  const int qt = blockIdx.x;
  const int g  = h / (H_NUM / G_NUM);
  const int tid = threadIdx.x;
  const int lane = tid & 31;
  const int w = tid >> 5;
  const int lrow = lane & 15;
  const int lhalf = lane >> 4;
  const int q0 = qt * 64;

  // Load the wave's 16x64 Q tile as two K=32 A-fragments (stays in VGPRs).
  FragU aq[2];
  {
    const unsigned short* qp = &Q[((size_t)h * S + q0 + w * 16 + lrow) * 64];
    load_fragA(aq[0], qp, lhalf);
    load_fragA(aq[1], qp + 32, lhalf);
  }

  f32x8 o_acc[4] = {};
  float m_i[8], l_i[8];
#pragma unroll
  for (int r = 0; r < 8; ++r) { m_i[r] = -3.0e38f; l_i[r] = 0.0f; }

  const float scale = 0.125f;  // HD^-0.5

  for (int kt = 0; kt <= qt; ++kt) {
    const int k0 = kt * 64;

    // K tile: async DMA straight to LDS (row-major, 512 uint4 / 128 threads).
#pragma unroll
    for (int i = 0; i < 4; ++i) {
      int idx = tid + i * 128;          // 0..511
      int r = idx >> 3, c = idx & 7;    // key row 0..63, 8-elem group 0..7
      async_g2l_b128(&Kk[((size_t)g * S + k0 + r) * 64 + c * 8], &Ks[r * 64 + c * 8]);
    }
    // V tile: load + 16-bit scatter transpose into Vt[hd][key].
#pragma unroll
    for (int i = 0; i < 4; ++i) {
      int idx = tid + i * 128;
      int r = idx >> 3, c = idx & 7;
      uint4 vv = *(const uint4*)&V[((size_t)g * S + k0 + r) * 64 + c * 8];
      const unsigned short* pv = (const unsigned short*)&vv;
#pragma unroll
      for (int e = 0; e < 8; ++e) Vt[(c * 8 + e) * 64 + r] = pv[e];
    }
    WAIT_ASYNCCNT(0);
    __syncthreads();

    // Scores: 4 x (16x16) tiles across 64 keys, K-loop over HD=64 (2 WMMAs).
    f32x8 sc[4];
#pragma unroll
    for (int n = 0; n < 4; ++n) {
      FragU bk0, bk1;
      const unsigned short* kp = &Ks[(n * 16 + lrow) * 64];
      load_fragB(bk0, kp, lhalf);
      load_fragB(bk1, kp + 32, lhalf);
      f32x8 s = {};
      s = wmma_bf16(aq[0], bk0, s);
      s = wmma_bf16(aq[1], bk1, s);
      sc[n] = s;
    }

    // Online softmax update (per row owned by this lane in C layout).
    const bool diag = (kt == qt);
#pragma unroll
    for (int r = 0; r < 8; ++r) {
      const int grow = q0 + w * 16 + r + 8 * lhalf;
      float rmax = -3.0e38f;
#pragma unroll
      for (int n = 0; n < 4; ++n) {
        float val = sc[n][r] * scale;
        if (diag && (k0 + n * 16 + lrow) > grow) val = -3.0e38f;
        sc[n][r] = val;
        rmax = fmaxf(rmax, val);
      }
#pragma unroll
      for (int off = 1; off < 16; off <<= 1)
        rmax = fmaxf(rmax, __shfl_xor(rmax, off, 32));
      float mnew = fmaxf(m_i[r], rmax);
      float alpha = __expf(m_i[r] - mnew);
      m_i[r] = mnew;
      float rsum = 0.0f;
#pragma unroll
      for (int n = 0; n < 4; ++n) {
        float p = __expf(sc[n][r] - mnew);
        sc[n][r] = p;
        rsum += p;
      }
#pragma unroll
      for (int off = 1; off < 16; off <<= 1)
        rsum += __shfl_xor(rsum, off, 32);
      l_i[r] = l_i[r] * alpha + rsum;
#pragma unroll
      for (int n = 0; n < 4; ++n) o_acc[n][r] *= alpha;
    }

    // P (C layout) -> LDS row-major 16x64 so it can be re-read as A-fragments.
#pragma unroll
    for (int n = 0; n < 4; ++n)
#pragma unroll
      for (int r = 0; r < 8; ++r)
        Ps[w][(r + 8 * lhalf) * 64 + n * 16 + lrow] = f2bf(sc[n][r]);
    __syncthreads();

    // p @ v : A = P (16 x 64 keys), B = V^T fragments from Vt[hd][key].
    FragU ap0, ap1;
    {
      const unsigned short* pp = &Ps[w][lrow * 64];
      load_fragA(ap0, pp, lhalf);
      load_fragA(ap1, pp + 32, lhalf);
    }
#pragma unroll
    for (int n = 0; n < 4; ++n) {
      FragU bv0, bv1;
      const unsigned short* vp = &Vt[(n * 16 + lrow) * 64];
      load_fragB(bv0, vp, lhalf);
      load_fragB(bv1, vp + 32, lhalf);
      o_acc[n] = wmma_bf16(ap0, bv0, o_acc[n]);
      o_acc[n] = wmma_bf16(ap1, bv1, o_acc[n]);
    }
    __syncthreads();
  }

  // Normalize and store ctx in (s, h*64+hd) bf16.
#pragma unroll
  for (int r = 0; r < 8; ++r) {
    const int grow = q0 + w * 16 + r + 8 * lhalf;
    const float rinv = 1.0f / l_i[r];
#pragma unroll
    for (int n = 0; n < 4; ++n) {
      float val = o_acc[n][r] * rinv;
      O[(size_t)grow * ldo + h * 64 + n * 16 + lrow] = f2bf(val);
    }
  }
}

// ---------------------------------------------------------------------------
// Host-side orchestration
// ---------------------------------------------------------------------------
extern "C" void kernel_launch(void* const* d_in, const int* in_sizes, int n_in,
                              void* d_out, int out_size, void* d_ws, size_t ws_size,
                              hipStream_t stream) {
  const float* x    = (const float*)d_in[0];
  // d_in[1] = mask (bool) -- causal mask recomputed on device, unused.
  const float* cosT = (const float*)d_in[2];
  const float* sinT = (const float*)d_in[3];
  const float* wq   = (const float*)d_in[4];
  const float* wk   = (const float*)d_in[5];
  const float* wv   = (const float*)d_in[6];
  const float* wo   = (const float*)d_in[7];
  const float* qg   = (const float*)d_in[8];
  const float* kg   = (const float*)d_in[9];
  float* out = (float*)d_out;

  char* cur = (char*)d_ws;
  auto alloc = [&](size_t bytes) -> char* {
    char* r = cur;
    cur += (bytes + 255) & ~(size_t)255;
    return r;
  };

  unsigned short* xbf   = (unsigned short*)alloc((size_t)S_DIM * D_DIM * 2);
  unsigned short* wqbf  = (unsigned short*)alloc((size_t)HQ_DIM * D_DIM * 2);
  unsigned short* wkbf  = (unsigned short*)alloc((size_t)HK_DIM * D_DIM * 2);
  unsigned short* wvbf  = (unsigned short*)alloc((size_t)HK_DIM * D_DIM * 2);
  unsigned short* wobf  = (unsigned short*)alloc((size_t)D_DIM * HQ_DIM * 2);
  float*          qf    = (float*)alloc((size_t)S_DIM * HQ_DIM * 4);
  float*          kf    = (float*)alloc((size_t)S_DIM * HK_DIM * 4);
  float*          vf    = (float*)alloc((size_t)S_DIM * HK_DIM * 4);
  unsigned short* qbf   = (unsigned short*)alloc((size_t)H_NUM * S_DIM * HD_DIM * 2);
  unsigned short* kbf   = (unsigned short*)alloc((size_t)G_NUM * S_DIM * HD_DIM * 2);
  unsigned short* vbf   = (unsigned short*)alloc((size_t)G_NUM * S_DIM * HD_DIM * 2);
  unsigned short* ctxbf = (unsigned short*)alloc((size_t)S_DIM * HQ_DIM * 2);

  // 1) fp32 -> bf16 conversions
  k_f32_to_bf16<<<1024, 256, 0, stream>>>(x,  xbf,  S_DIM * D_DIM);
  k_f32_to_bf16<<<1024, 256, 0, stream>>>(wq, wqbf, HQ_DIM * D_DIM);
  k_f32_to_bf16<<<512,  256, 0, stream>>>(wk, wkbf, HK_DIM * D_DIM);
  k_f32_to_bf16<<<512,  256, 0, stream>>>(wv, wvbf, HK_DIM * D_DIM);
  k_f32_to_bf16<<<1024, 256, 0, stream>>>(wo, wobf, D_DIM * HQ_DIM);

  // 2) QKV projections (WMMA GEMM, 128x128 tiles)
  {
    dim3 gq(HQ_DIM / 128, S_DIM / 128);
    k_gemm_bf16_nt<<<gq, 256, 0, stream>>>(xbf, wqbf, qf, S_DIM, HQ_DIM, D_DIM);
    dim3 gk(HK_DIM / 128, S_DIM / 128);
    k_gemm_bf16_nt<<<gk, 256, 0, stream>>>(xbf, wkbf, kf, S_DIM, HK_DIM, D_DIM);
    k_gemm_bf16_nt<<<gk, 256, 0, stream>>>(xbf, wvbf, vf, S_DIM, HK_DIM, D_DIM);
  }

  // 3) RMSNorm + RoPE + transpose to (h,s,hd) bf16
  {
    dim3 gq(S_DIM, H_NUM);
    k_norm_rope_tr<<<gq, 64, 0, stream>>>(qf, qbf, qg, cosT, sinT, H_NUM, S_DIM, 1);
    dim3 gk(S_DIM, G_NUM);
    k_norm_rope_tr<<<gk, 64, 0, stream>>>(kf, kbf, kg, cosT, sinT, G_NUM, S_DIM, 1);
    k_norm_rope_tr<<<gk, 64, 0, stream>>>(vf, vbf, (const float*)nullptr, cosT, sinT,
                                          G_NUM, S_DIM, 0);
  }

  // 4) Causal flash attention (WMMA)
  {
    dim3 ga(S_DIM / 64, H_NUM);
    k_flash_attn<<<ga, 128, 0, stream>>>(qbf, kbf, vbf, ctxbf, S_DIM, HQ_DIM);
  }

  // 5) Output projection (WMMA GEMM) -> fp32 result
  {
    dim3 go(D_DIM / 128, S_DIM / 128);
    k_gemm_bf16_nt<<<go, 256, 0, stream>>>(ctxbf, wobf, out, S_DIM, D_DIM, HQ_DIM);
  }
}
